// KNNAttention_12034498363998
// MI455X (gfx1250) — matrix operand order repaired and checked by reading, not compiled, not run on hardware
//
#include <hip/hip_runtime.h>
#include <hip/hip_bf16.h>

typedef __attribute__((ext_vector_type(16))) _Float16 v16h;
typedef __attribute__((ext_vector_type(8)))  _Float16 v8h;
typedef __attribute__((ext_vector_type(8)))  float    v8f;
typedef __attribute__((ext_vector_type(4)))  float    v4f;
typedef __attribute__((ext_vector_type(2)))  float    v2f;

#define NBATCH 2
#define NHEADS 8
#define NSEQ   2048
#define NDIM   64
#define NKNN   32
#define NEG_BIG 3.4028234663852886e38f
#define WAVES_PER_BLOCK 4

// ---------------- preprocessing: kn = l2norm(k) as f16 row-major -------------
__global__ __launch_bounds__(256) void prep_k_kernel(const float* __restrict__ K,
                                                     _Float16* __restrict__ KN) {
  const int row = blockIdx.x * 256 + threadIdx.x;          // B*S rows
  if (row >= NBATCH * NSEQ) return;
  const float* src = K + (size_t)row * NDIM;
  float buf[NDIM];
  float ss = 0.f;
#pragma unroll
  for (int d = 0; d < NDIM; d += 4) {
    v4f a = *(const v4f*)(src + d);
#pragma unroll
    for (int j = 0; j < 4; ++j) { buf[d + j] = a[j]; ss += a[j] * a[j]; }
  }
  const float inv = 1.0f / fmaxf(sqrtf(ss), 1e-12f);
  _Float16* dst = KN + (size_t)row * NDIM;
#pragma unroll
  for (int g = 0; g < NDIM; g += 8) {
    v8h h;
#pragma unroll
    for (int j = 0; j < 8; ++j) h[j] = (_Float16)(buf[g + j] * inv);
    *(v8h*)(dst + g) = h;
  }
}

// ---------------- preprocessing: vt = V^T as f16 [b][d][s] -------------------
__global__ __launch_bounds__(256) void prep_v_kernel(const float* __restrict__ V,
                                                     _Float16* __restrict__ VT) {
  const int gid = blockIdx.x * 256 + threadIdx.x;          // B*D*(S/8)
  if (gid >= NBATCH * NDIM * (NSEQ / 8)) return;
  const int sc = gid & (NSEQ / 8 - 1);
  const int d  = (gid >> 8) & (NDIM - 1);
  const int b  = gid >> 14;
  const int s0 = sc * 8;
  v8h h;
#pragma unroll
  for (int i = 0; i < 8; ++i)
    h[i] = (_Float16)V[((size_t)b * NSEQ + s0 + i) * NDIM + d];
  *(v8h*)(VT + ((size_t)b * NDIM + d) * NSEQ + s0) = h;
}

// ---------------- main kernel ------------------------------------------------
// Each wave: 16 queries of one (b,h). Flash-attention over causal keys with
// v_wmma_f32_16x16x32_f16 (S^T orientation: lane = query), mem-KV folded in
// as the initial online-softmax state. PRE=true uses preprocessed f16 K/V^T.
template <bool PRE>
__global__ __launch_bounds__(WAVES_PER_BLOCK * 32) void knn_attn_kernel(
    const float* __restrict__ Q, const float* __restrict__ K,
    const float* __restrict__ V, const float* __restrict__ SCALEP,
    const float* __restrict__ MEMKV, const float* __restrict__ MASK,
    const unsigned char* __restrict__ MEMMASK,
    const _Float16* __restrict__ KN, const _Float16* __restrict__ VT,
    float* __restrict__ OUT)
{
  __shared__ float sq[WAVES_PER_BLOCK][16][NDIM];   // normalized q rows (f32)
  __shared__ float so[WAVES_PER_BLOCK][16][NDIM];   // mem-KV output seed
  __shared__ float smx[WAVES_PER_BLOCK][16];        // mem-KV running max
  __shared__ float ssum[WAVES_PER_BLOCK][16];       // mem-KV running sum

  const int wave = threadIdx.x >> 5;
  const int lane = threadIdx.x & 31;
  const int qq   = lane & 15;     // query column (C-layout N)
  const int half = lane >> 4;     // 0: M rows 0..7 / K lo ; 1: M rows 8..15 / K hi

  const int tile = blockIdx.x * WAVES_PER_BLOCK + wave;   // 2048 tiles total
  const int t  = tile & (NSEQ / 16 - 1);
  const int h  = (tile >> 7) & (NHEADS - 1);
  const int b  = tile >> 10;
  const int q0 = t * 16;
  const int bh = b * NHEADS + h;

  const float scale = __expf(SCALEP[h]);

  // ---------- load + l2norm Q tile; build B fragments (Qn^T) ----------
  const float* qrow = Q + ((size_t)bh * NSEQ + (q0 + qq)) * NDIM;
  const int d0 = half * 8;
  float s0[16], s1[16];
#pragma unroll
  for (int i = 0; i < 4; ++i) {
    v4f a = *(const v4f*)(qrow + d0 + 4 * i + (i & 2) * 4);
    v4f c = *(const v4f*)(qrow + d0 + 32 + 4 * i + (i & 2) * 4);
#pragma unroll
    for (int j = 0; j < 4; ++j) { s0[4 * i + j] = a[j]; s1[4 * i + j] = c[j]; }
  }
  float ssq = 0.f;
#pragma unroll
  for (int i = 0; i < 16; ++i) ssq += s0[i] * s0[i] + s1[i] * s1[i];
  ssq += __shfl_xor(ssq, 16);
  const float qinv = 1.0f / fmaxf(sqrtf(ssq), 1e-12f);
  v16h bq0, bq1;   // B-layout: lane half h -> dims c*32+{8h..8h+7, 16+8h..16+8h+7}
#pragma unroll
  for (int i = 0; i < 16; ++i) {
    s0[i] *= qinv; s1[i] *= qinv;
    bq0[i] = (_Float16)s0[i];
    bq1[i] = (_Float16)s1[i];
    sq[wave][qq][d0 + i + (i & 8)]      = s0[i];
    sq[wave][qq][d0 + 32 + i + (i & 8)] = s1[i];
  }
  __syncthreads();

  // ---------- memory-KV phase: stream mem_kv exactly once ----------
  for (int q2 = 0; q2 < 16; ++q2) {
    const int qi2 = q0 + q2;
    const float* mk = MEMKV + ((((size_t)bh * NSEQ + qi2) * NKNN + lane) * 2) * NDIM;
    if (q2 < 15) __builtin_prefetch(mk + (size_t)NKNN * 2 * NDIM, 0, 0);
    float dot = 0.f;
#pragma unroll
    for (int d = 0; d < NDIM; d += 4) {
      v4f a = *(const v4f*)&sq[wave][q2][d];
      v4f c = *(const v4f*)(mk + d);
      dot += a[0] * c[0] + a[1] * c[1] + a[2] * c[2] + a[3] * c[3];
    }
    const unsigned char mm = MEMMASK[((size_t)bh * NSEQ + qi2) * NKNN + lane];
    float logit = mm ? dot * scale : -NEG_BIG;
    float mx = logit;
#pragma unroll
    for (int off = 16; off > 0; off >>= 1) mx = fmaxf(mx, __shfl_xor(mx, off));
    float p = mm ? __expf(logit - mx) : 0.f;   // explicit 0 if fully masked
    float sm = p;
#pragma unroll
    for (int off = 16; off > 0; off >>= 1) sm += __shfl_xor(sm, off);
    if (lane == 0) { smx[wave][q2] = mx; ssum[wave][q2] = sm; }
    // weighted mem_v sum: lanes switch to dim-parallel (2 dims/lane)
    float vo0 = 0.f, vo1 = 0.f;
    const float* mvbase = MEMKV + ((((size_t)bh * NSEQ + qi2) * NKNN) * 2 + 1) * NDIM;
#pragma unroll 8
    for (int jm = 0; jm < NKNN; ++jm) {
      float pj = __shfl(p, jm);
      v2f vv = *(const v2f*)(mvbase + (size_t)jm * 2 * NDIM + 2 * lane);
      vo0 += pj * vv[0];
      vo1 += pj * vv[1];
    }
    v2f w; w[0] = vo0; w[1] = vo1;
    *(v2f*)&so[wave][q2][2 * lane] = w;
  }
  __syncthreads();

  // ---------- seed flash state from mem-KV phase ----------
  float m = smx[wave][qq];
  float l = ssum[wave][qq];
  v8f acc[4];   // out tile 16q x 64d: C-layout lane=dim, vgpr=q row
#pragma unroll
  for (int f = 0; f < 4; ++f)
#pragma unroll
    for (int r = 0; r < 8; ++r)
      acc[f][r] = so[wave][r + half * 8][f * 16 + qq];

  const float* kbase = K + (size_t)b * NSEQ * NDIM;
  const float* vbase = V + (size_t)b * NSEQ * NDIM;
  const float* mrow  = MASK + (size_t)b * NSEQ;
  const int qi_l = q0 + qq;
  const int kend = q0 + 16;

  // ---------- causal flash loop over 32-key blocks ----------
  for (int kb = 0; kb < kend; kb += 32) {
    // Kn A-fragments: 2 key blocks x 2 dim chunks
    v16h ak[2][2];
#pragma unroll
    for (int blk = 0; blk < 2; ++blk) {
      if constexpr (PRE) {
        const _Float16* krow = KN + (size_t)(b * NSEQ + kb + blk * 16 + qq) * NDIM;
#pragma unroll
        for (int c = 0; c < 2; ++c) {
          const int base = c * 32 + half * 8;   // A-layout: {g..g+7, g+16..g+23}
          v8h lo = *(const v8h*)(krow + base);
          v8h hi = *(const v8h*)(krow + base + 16);
#pragma unroll
          for (int j = 0; j < 8; ++j) { ak[blk][c][j] = lo[j]; ak[blk][c][8 + j] = hi[j]; }
        }
      } else {
        const float* krow = kbase + (size_t)(kb + blk * 16 + qq) * NDIM;
        float kf[2][16];
#pragma unroll
        for (int c = 0; c < 2; ++c) {
          const int base = c * 32 + half * 8;
#pragma unroll
          for (int i = 0; i < 2; ++i) {
            v4f a = *(const v4f*)(krow + base + 4 * i);
            v4f d = *(const v4f*)(krow + base + 16 + 4 * i);
#pragma unroll
            for (int j = 0; j < 4; ++j) { kf[c][4 * i + j] = a[j]; kf[c][8 + 4 * i + j] = d[j]; }
          }
        }
        float kss = 0.f;
#pragma unroll
        for (int c = 0; c < 2; ++c)
#pragma unroll
          for (int i = 0; i < 16; ++i) kss += kf[c][i] * kf[c][i];
        kss += __shfl_xor(kss, 16);
        const float kinv = 1.0f / fmaxf(sqrtf(kss), 1e-12f);
#pragma unroll
        for (int c = 0; c < 2; ++c)
#pragma unroll
          for (int i = 0; i < 16; ++i) ak[blk][c][i] = (_Float16)(kf[c][i] * kinv);
      }
    }

    // S^T = Kn * Qn^T  (lane = query, vgpr = key)
    v8f z = {};
    v8f st0 = __builtin_amdgcn_wmma_f32_16x16x32_f16(false, ak[0][0], false, bq0, (short)0, z,   false, false);
    st0     = __builtin_amdgcn_wmma_f32_16x16x32_f16(false, ak[0][1], false, bq1, (short)0, st0, false, false);
    v8f st1 = __builtin_amdgcn_wmma_f32_16x16x32_f16(false, ak[1][0], false, bq0, (short)0, z,   false, false);
    st1     = __builtin_amdgcn_wmma_f32_16x16x32_f16(false, ak[1][1], false, bq1, (short)0, st1, false, false);

    // key-mask values for this lane's rows (vectorized)
    v4f mk0a = *(const v4f*)(mrow + kb + half * 8);
    v4f mk0b = *(const v4f*)(mrow + kb + half * 8 + 4);
    v4f mk1a = *(const v4f*)(mrow + kb + 16 + half * 8);
    v4f mk1b = *(const v4f*)(mrow + kb + 16 + half * 8 + 4);

    // logits + key-mask + causal + online softmax (per-lane = per-query)
    float lg[16];
#pragma unroll
    for (int r = 0; r < 8; ++r) {
      const int key0 = kb + r + half * 8;
      const int key1 = key0 + 16;
      const float mv0 = (r < 4) ? mk0a[r & 3] : mk0b[r & 3];
      const float mv1 = (r < 4) ? mk1a[r & 3] : mk1b[r & 3];
      float x0 = st0[r] * scale - NEG_BIG * (1.0f - mv0);
      float x1 = st1[r] * scale - NEG_BIG * (1.0f - mv1);
      if (key0 > qi_l) x0 = -NEG_BIG;
      if (key1 > qi_l) x1 = -NEG_BIG;
      lg[r] = x0; lg[8 + r] = x1;
    }
    float mloc = lg[0];
#pragma unroll
    for (int i = 1; i < 16; ++i) mloc = fmaxf(mloc, lg[i]);
    mloc = fmaxf(mloc, __shfl_xor(mloc, 16));
    const float mnew = fmaxf(m, mloc);
    float pr[16]; float sloc = 0.f;
#pragma unroll
    for (int i = 0; i < 16; ++i) { pr[i] = __expf(lg[i] - mnew); sloc += pr[i]; }
    sloc += __shfl_xor(sloc, 16);
    const float alpha = __expf(m - mnew);
    l = l * alpha + sloc;
    m = mnew;
    float af[8];
#pragma unroll
    for (int r = 0; r < 8; ++r) af[r] = __shfl(alpha, r + half * 8);
#pragma unroll
    for (int f = 0; f < 4; ++f)
#pragma unroll
      for (int r = 0; r < 8; ++r) acc[f][r] *= af[r];

    // P lands exactly in A-fragment layout: no cross-lane exchange needed
    v16h pa;
#pragma unroll
    for (int i = 0; i < 16; ++i) pa[i] = (_Float16)pr[i];

    // V B-fragments + PV accumulation
#pragma unroll
    for (int f = 0; f < 4; ++f) {
      const int d = f * 16 + qq;
      v16h bv;
      if constexpr (PRE) {
        const _Float16* vtrow = VT + ((size_t)b * NDIM + d) * NSEQ + kb + half * 8;
        v8h lo = *(const v8h*)(vtrow);
        v8h hi = *(const v8h*)(vtrow + 16);
#pragma unroll
        for (int j = 0; j < 8; ++j) { bv[j] = lo[j]; bv[8 + j] = hi[j]; }
      } else {
#pragma unroll
        for (int r = 0; r < 8; ++r) {
          const int key = kb + 2 * r + half * 8 + (r & 4) * 2;
          bv[2 * r]     = (_Float16)vbase[(size_t)key * NDIM + d];
          bv[2 * r + 1] = (_Float16)vbase[(size_t)(key + 1) * NDIM + d];
        }
      }
      acc[f] = __builtin_amdgcn_wmma_f32_16x16x32_f16(false, pa, false, bv, (short)0, acc[f], false, false);
    }
  }

  // ---------- normalize + store ----------
  float linv[8];
#pragma unroll
  for (int r = 0; r < 8; ++r) linv[r] = 1.0f / __shfl(l, r + half * 8);
#pragma unroll
  for (int f = 0; f < 4; ++f)
#pragma unroll
    for (int r = 0; r < 8; ++r)
      OUT[((size_t)bh * NSEQ + q0 + r + half * 8) * NDIM + f * 16 + qq] =
          acc[f][r] * linv[r];
}

extern "C" void kernel_launch(void* const* d_in, const int* in_sizes, int n_in,
                              void* d_out, int out_size, void* d_ws, size_t ws_size,
                              hipStream_t stream) {
  const float* q     = (const float*)d_in[0];
  const float* k     = (const float*)d_in[1];
  const float* v     = (const float*)d_in[2];
  const float* sp    = (const float*)d_in[3];
  const float* mkv   = (const float*)d_in[4];
  const float* mask  = (const float*)d_in[5];
  const unsigned char* mm = (const unsigned char*)d_in[6];
  float* out = (float*)d_out;

  const int tiles  = NBATCH * NHEADS * (NSEQ / 16);          // 2048
  const int blocks = tiles / WAVES_PER_BLOCK;                // 512

  const size_t elems = (size_t)NBATCH * NSEQ * NDIM;         // 262144
  const size_t need  = 2 * elems * sizeof(_Float16);         // 1 MB

  if (d_ws != nullptr && ws_size >= need) {
    _Float16* kn = (_Float16*)d_ws;
    _Float16* vt = kn + elems;
    prep_k_kernel<<<(NBATCH * NSEQ + 255) / 256, 256, 0, stream>>>(k, kn);
    prep_v_kernel<<<(NBATCH * NDIM * (NSEQ / 8) + 255) / 256, 256, 0, stream>>>(v, vt);
    knn_attn_kernel<true><<<blocks, WAVES_PER_BLOCK * 32, 0, stream>>>(
        q, k, v, sp, mkv, mask, mm, kn, vt, out);
  } else {
    knn_attn_kernel<false><<<blocks, WAVES_PER_BLOCK * 32, 0, stream>>>(
        q, k, v, sp, mkv, mask, mm, nullptr, nullptr, out);
  }
}
